// Stack_65944927862836
// MI455X (gfx1250) — matrix operand order repaired and checked by reading, not compiled, not run on hardware
//
#include <hip/hip_runtime.h>

// Problem constants from the reference.
constexpr int T = 512;
constexpr int B = 128;
constexpr int E = 256;

typedef __attribute__((ext_vector_type(16))) _Float16     v16h;
typedef __attribute__((ext_vector_type(8)))  float        v8f;
typedef __attribute__((ext_vector_type(4)))  unsigned int u32x4;

// ---------------------------------------------------------------------------
// Phase A: per-batch sequential stack scan producing the read-weight matrix
// W[b][t][slot] in f16 (zero above the diagonal). One wave32 per batch column,
// 16 slots per lane, suffix sums via lane-local adds + wave shfl scan.
//
// Closed forms (S_i = suffix sum of old strengths):
//   pop(u):  s'_i = max(S_i-u,0) - max(S_{i+1}-u,0);  s'_t = d
//   S'_i   = max(S_i-u,0) + d        (for i <= t; else 0)
//   w_i    = min(S'_i,1) - min(S'_{i+1},1)
// ---------------------------------------------------------------------------
__global__ __launch_bounds__(32) void stack_scan_weights(
    const float* __restrict__ ug,   // [T,B]
    const float* __restrict__ dg,   // [T,B]
    _Float16* __restrict__ Wh)      // [B,T,T] f16 workspace
{
  const int b    = blockIdx.x;
  const int lane = threadIdx.x & 31;

  float s[16];
#pragma unroll
  for (int j = 0; j < 16; ++j) s[j] = 0.0f;

  _Float16* __restrict__ Wb = Wh + (size_t)b * T * T;
  const int base = lane * 16;

  for (int t = 0; t < T; ++t) {
    const float u = ug[t * B + b];
    const float d = dg[t * B + b];

    // Lane-local suffix sums: P[j] = sum_{j'>=j} s[j'],  P[16] = 0.
    float P[17];
    P[16] = 0.0f;
#pragma unroll
    for (int j = 15; j >= 0; --j) P[j] = P[j + 1] + s[j];

    // Wave-wide inclusive suffix scan of lane totals (no LDS, no barrier).
    float acc = P[0];
#pragma unroll
    for (int off = 1; off < 32; off <<= 1) {
      float o = __shfl_down(acc, off, 32);
      acc += (lane + off < 32) ? o : 0.0f;
    }
    const float excl = acc - P[0];  // sum over lanes strictly above this lane

    // A[j] = max(S_i - u, 0) at slot i = base + j  (A[16] = next lane's first).
    float A[17];
#pragma unroll
    for (int j = 0; j <= 16; ++j) A[j] = fmaxf(excl + P[j] - u, 0.0f);

    union { _Float16 h[16]; u32x4 q[2]; } wrow;
#pragma unroll
    for (int j = 0; j < 16; ++j) {
      const int i = base + j;
      float sn = A[j] - A[j + 1];       // pop result (0 for i >= t)
      if (i == t) sn = d;               // push
      s[j] = sn;
      const float m0 = (i     <= t) ? fminf(A[j]     + d, 1.0f) : 0.0f;
      const float m1 = (i + 1 <= t) ? fminf(A[j + 1] + d, 1.0f) : 0.0f;
      wrow.h[j] = (_Float16)(m0 - m1);  // read weight for slot i at step t
    }

    // Coalesced: lane L writes halves [16L,16L+16) of row t -> 2 x b128.
    u32x4* dst = (u32x4*)(Wb + (size_t)t * T + base);
    dst[0] = wrow.q[0];
    dst[1] = wrow.q[1];
  }
}

// ---------------------------------------------------------------------------
// Phase B: R_b = W_b @ V_b using V_WMMA_F32_16X16X32_F16, f32 accumulate.
// Grid (E/128, T/64, B), 256 threads = 8 waves arranged 2(M) x 4(N).
// Each wave owns a 32x32 output tile: 2 A frags x 2 B frags -> 4 WMMAs with
// 2x fragment reuse (1.5 KB of L2 loads per WMMA). Triangular: k runs only
// to t0+64 (W is zero above the diagonal, so in-tile overshoot is exact).
//
// A 16x32 f16 fragment (ISA layout): lanes 0-15 row M=l, K = kb+{0..7},
// kb+16+{0..7}; lanes 16-31 same rows, K shifted by +8 -> two aligned b128
// loads per lane from the f16 W rows.
// B 32x16 f16 fragment: lane n (0-15) holds column n, K = kb+{0..7}, kb+16+{0..7}; lanes
// 16-31 same columns, K shifted by +8 -> 16 strided f32 loads of v, converted to f16
// (each load is 2x64B fully-used segments across the wave).
// ---------------------------------------------------------------------------
__global__ __launch_bounds__(256) void stack_read_gemm(
    const float* __restrict__ vg,      // [T,B,E] f32
    const _Float16* __restrict__ Wh,   // [B,T,T] f16
    float* __restrict__ outg)          // [T,B,E] f32
{
  const int b    = blockIdx.z;
  const int t0   = blockIdx.y * 64;
  const int e0   = blockIdx.x * 128;
  const int lane = threadIdx.x & 31;
  const int wid  = threadIdx.x >> 5;

  const int tRow = t0 + (wid & 1) * 32;   // 2 waves along M, 32 rows each
  const int eCol = e0 + (wid >> 1) * 32;  // 4 waves along N, 32 cols each
  const int half = lane >> 4;             // 0: lanes 0-15, 1: lanes 16-31
  const int l15  = lane & 15;

  const _Float16* __restrict__ Arow0 =
      Wh + ((size_t)b * T + (tRow + l15)) * T + half * 8;
  const _Float16* __restrict__ Arow1 = Arow0 + (size_t)16 * T;
  const float* __restrict__ Bbase = vg + (size_t)b * E;

  v8f acc00 = {};   // rows tRow+0..15,  cols eCol+0..15
  v8f acc01 = {};   // rows tRow+0..15,  cols eCol+16..31
  v8f acc10 = {};   // rows tRow+16..31, cols eCol+0..15
  v8f acc11 = {};   // rows tRow+16..31, cols eCol+16..31

  const int kmax = t0 + 64;  // rows in this tile never read k beyond t0+63
  for (int kb = 0; kb < kmax; kb += 32) {
    // ---- A fragments: two aligned 16B loads each of consecutive f16 ------
    union { u32x4 q[2]; v16h h; } au0, au1;
    au0.q[0] = *(const u32x4*)(Arow0 + kb);        // K = kb + 8*half + {0..7}
    au0.q[1] = *(const u32x4*)(Arow0 + kb + 16);   // K = kb+16+8*half + {0..7}
    au1.q[0] = *(const u32x4*)(Arow1 + kb);
    au1.q[1] = *(const u32x4*)(Arow1 + kb + 16);
    const v16h a0 = au0.h;
    const v16h a1 = au1.h;

    // ---- B fragments: gather 16 slots of v for this lane's column --------
    const int k2 = kb + half * 8;
    const float* p0 = Bbase + (size_t)k2 * (B * E) + (eCol + l15);
    const float* p1 = p0 + 16;
    v16h b0, b1;
#pragma unroll
    for (int j = 0; j < 16; ++j) {
      b0[j] = (_Float16)p0[(size_t)(j < 8 ? j : j + 8) * (B * E)];
      b1[j] = (_Float16)p1[(size_t)(j < 8 ? j : j + 8) * (B * E)];
    }

    acc00 = __builtin_amdgcn_wmma_f32_16x16x32_f16(
        false, a0, false, b0, (short)0, acc00, false, false);
    acc01 = __builtin_amdgcn_wmma_f32_16x16x32_f16(
        false, a0, false, b1, (short)0, acc01, false, false);
    acc10 = __builtin_amdgcn_wmma_f32_16x16x32_f16(
        false, a1, false, b0, (short)0, acc10, false, false);
    acc11 = __builtin_amdgcn_wmma_f32_16x16x32_f16(
        false, a1, false, b1, (short)0, acc11, false, false);
  }

  // ---- C/D store: VGPR r -> M = 8*half + r, N = l15 ----
  const int mBase = tRow + half * 8;
  const int nc0   = eCol + l15;
  float* __restrict__ orow = outg + (size_t)b * E;
#pragma unroll
  for (int r = 0; r < 8; ++r) {
    const size_t row0 = (size_t)(mBase + r) * B * E;        // upper 16 rows
    const size_t row1 = (size_t)(mBase + 16 + r) * B * E;   // lower 16 rows
    orow[row0 + nc0]      = acc00[r];
    orow[row0 + nc0 + 16] = acc01[r];
    orow[row1 + nc0]      = acc10[r];
    orow[row1 + nc0 + 16] = acc11[r];
  }
}

// ---------------------------------------------------------------------------
// Inputs (setup_inputs order): d_in[0]=v [T,B,E] f32, d_in[1]=u [T,B] f32,
// d_in[2]=d [T,B] f32. Output: rs [T,B,E] f32. Workspace: 64 MB f16 weights.
// ---------------------------------------------------------------------------
extern "C" void kernel_launch(void* const* d_in, const int* in_sizes, int n_in,
                              void* d_out, int out_size, void* d_ws, size_t ws_size,
                              hipStream_t stream) {
  (void)in_sizes; (void)n_in; (void)out_size; (void)ws_size;
  const float* v = (const float*)d_in[0];
  const float* u = (const float*)d_in[1];
  const float* d = (const float*)d_in[2];
  float* out = (float*)d_out;
  _Float16* Wh = (_Float16*)d_ws;  // needs B*T*T*2 = 64 MB

  stack_scan_weights<<<dim3(B), dim3(32), 0, stream>>>(u, d, Wh);
  stack_read_gemm<<<dim3(E / 128, T / 64, B), dim3(256), 0, stream>>>(v, Wh, out);
}
